// SelfAttention_3332894621986
// MI455X (gfx1250) — matrix operand order repaired and checked by reading, not compiled, not run on hardware
//
#include <hip/hip_runtime.h>

typedef __attribute__((ext_vector_type(16))) _Float16 v16h;
typedef __attribute__((ext_vector_type(8)))  _Float16 v8h;
typedef __attribute__((ext_vector_type(8)))  float    v8f;
typedef __attribute__((ext_vector_type(4)))  float    v4f;

#define BB 4
#define CC 512
#define TT 4096
#define EE 512

union ABfrag { v16h v; v8h h[2]; };

__device__ __forceinline__ v8f wmma16(v16h a, v16h b, v8f c) {
  // 8 args: (neg_a, A, neg_b, B, c_mod, C, reuse_a, reuse_b)
  return __builtin_amdgcn_wmma_f32_16x16x32_f16(false, a, false, b, (short)0, c,
                                                false, false);
}

// ---------------------------------------------------------------------------
// Kernel 0: convert Wq, Wk, Wv, W2 (each 512x512 f32) -> f16 packed in ws
// ---------------------------------------------------------------------------
__global__ __launch_bounds__(256) void cvt_weights(const float* __restrict__ Wq,
                                                   const float* __restrict__ Wk,
                                                   const float* __restrict__ Wv,
                                                   const float* __restrict__ W2,
                                                   _Float16* __restrict__ Wh) {
  int g = blockIdx.x * 256 + threadIdx.x;      // 4 * 262144 total
  int sel = g >> 18;
  int off = g & 262143;
  const float* src = (sel == 0) ? Wq : (sel == 1) ? Wk : (sel == 2) ? Wv : W2;
  Wh[g] = (_Float16)src[off];
}

// ---------------------------------------------------------------------------
// Kernel 1: fused QKV projection.
//   A = x^T tile staged to LDS (transpose f32->f16), B = weight rows (direct
//   global b128 loads, K contiguous). V stored transposed [B,E,T] so the
//   attention PV B-fragments become contiguous loads.
// Block tile: 128 (T) x 64 (E-chunk), 8 waves = 4(m) x 2(n), wave tile 32x32.
// grid = (T/128, B, 3 * E/64)
// ---------------------------------------------------------------------------
__global__ __launch_bounds__(256) void qkv_gemm(
    const float* __restrict__ x, const _Float16* __restrict__ Wh,
    const float* __restrict__ bq, const float* __restrict__ bk,
    const float* __restrict__ bv, _Float16* __restrict__ qh,
    _Float16* __restrict__ kh, _Float16* __restrict__ vth) {
  __shared__ __align__(16) _Float16 Xs[128][40];   // [t][c], padded stride

  const int t0     = blockIdx.x * 128;
  const int b      = blockIdx.y;
  const int which  = blockIdx.z / 8;   // 0=q 1=k 2=v
  const int echunk = blockIdx.z % 8;
  const int tid  = threadIdx.x;
  const int lane = tid & 31;
  const int wave = tid >> 5;
  const int wm = wave & 3;             // 32 rows each
  const int wn = wave >> 2;            // 32 cols each
  const int lrow = lane & 15;
  const int kbA = (lane < 16) ? 0 : 8;
  const int kbB = (lane < 16) ? 0 : 8;
  const int hi8 = (lane < 16) ? 0 : 8;

  const _Float16* W = Wh + (size_t)which * EE * CC;
  const float* bias = (which == 0) ? bq : (which == 1) ? bk : bv;
  const float* xb = x + (size_t)b * CC * TT;

  v8f acc[2][2];
  for (int i = 0; i < 2; ++i)
    for (int j = 0; j < 2; ++j) acc[i][j] = (v8f)0.0f;

  for (int k0 = 0; k0 < CC; k0 += 32) {
    __syncthreads();
    // stage 32(c) x 128(t) f32 -> f16, transposed into Xs[t][c]
    for (int p = 0; p < 4; ++p) {
      int f  = tid + p * 256;          // float4 unit index, 0..1023
      int c  = f >> 5;                 // 0..31
      int t4 = (f & 31) * 4;           // 0..124
      v4f d = *(const v4f*)(xb + (size_t)(k0 + c) * TT + t0 + t4);
      Xs[t4 + 0][c] = (_Float16)d[0];
      Xs[t4 + 1][c] = (_Float16)d[1];
      Xs[t4 + 2][c] = (_Float16)d[2];
      Xs[t4 + 3][c] = (_Float16)d[3];
    }
    __syncthreads();

    ABfrag A[2];
    for (int mi = 0; mi < 2; ++mi) {
      const _Float16* ap = &Xs[wm * 32 + mi * 16 + lrow][kbA];
      A[mi].h[0] = *(const v8h*)ap;
      A[mi].h[1] = *(const v8h*)(ap + 16);
    }
    for (int ni = 0; ni < 2; ++ni) {
      int col = echunk * 64 + wn * 32 + ni * 16 + lrow;
      const _Float16* bp = W + (size_t)col * CC + k0 + kbB;
      ABfrag Bf;
      Bf.h[0] = *(const v8h*)bp;
      Bf.h[1] = *(const v8h*)(bp + 16);
      for (int mi = 0; mi < 2; ++mi)
        acc[mi][ni] = wmma16(A[mi].v, Bf.v, acc[mi][ni]);
    }
  }

  // epilogue: + bias, cvt f16, scatter (v goes out transposed)
  for (int ni = 0; ni < 2; ++ni) {
    int col = echunk * 64 + wn * 32 + ni * 16 + lrow;
    float bval = bias[col];
    for (int mi = 0; mi < 2; ++mi) {
      for (int r = 0; r < 8; ++r) {
        int trow = t0 + wm * 32 + mi * 16 + r + hi8;
        _Float16 hv = (_Float16)(acc[mi][ni][r] + bval);
        if (which == 0)
          qh[((size_t)b * TT + trow) * EE + col] = hv;
        else if (which == 1)
          kh[((size_t)b * TT + trow) * EE + col] = hv;
        else
          vth[((size_t)b * EE + col) * TT + trow] = hv;
      }
    }
  }
}

// ---------------------------------------------------------------------------
// Kernel 2: causal flash attention.
//   Block = 16 q-rows, 8 waves. Per 128-key tile: each wave computes one
//   16x16 S tile (K=E=512 -> 16 WMMA), causal mask, online softmax via
//   shfl_xor width-16 (matches C-matrix lane layout), P f16 in LDS, then
//   PV: each wave owns 64 E-columns (16 WMMA). V is [E,T] so B-frags are
//   contiguous. grid = (T/16, B).
// ---------------------------------------------------------------------------
__global__ __launch_bounds__(256) void attn_flash(
    const _Float16* __restrict__ qh, const _Float16* __restrict__ kh,
    const _Float16* __restrict__ vth, _Float16* __restrict__ ench) {
  __shared__ __align__(16) float    Sbuf[16][128];
  __shared__ __align__(16) _Float16 Pbuf[16][128];
  __shared__ __align__(16) float mstate[16];
  __shared__ __align__(16) float lstate[16];
  __shared__ __align__(16) float scl[16];

  const int q0 = blockIdx.x * 16;
  const int b  = blockIdx.y;
  const int tid  = threadIdx.x;
  const int lane = tid & 31;
  const int wave = tid >> 5;
  const int lrow = lane & 15;
  const int kbA = (lane < 16) ? 0 : 8;
  const int kbB = (lane < 16) ? 0 : 8;
  const int hi8 = (lane < 16) ? 0 : 8;

  const _Float16* qb = qh + (size_t)b * TT * EE;
  const _Float16* kb = kh + (size_t)b * TT * EE;
  const _Float16* vb = vth + (size_t)b * EE * TT;

  if (tid < 16) { mstate[tid] = -1e30f; lstate[tid] = 0.0f; }
  v8f oacc[4];
  for (int i = 0; i < 4; ++i) oacc[i] = (v8f)0.0f;
  __syncthreads();

  const int nkt = (q0 >> 7) + 1;       // causal key-tile count
  for (int kt = 0; kt < nkt; ++kt) {
    const int kt0 = kt * 128;
    const int keycol = kt0 + wave * 16 + lrow;

    // S = Q K^T (16x16 per wave)
    v8f s = (v8f)0.0f;
    for (int e0 = 0; e0 < EE; e0 += 32) {
      ABfrag A, Bf;
      const _Float16* ap = qb + (size_t)(q0 + lrow) * EE + e0 + kbA;
      A.h[0] = *(const v8h*)ap;
      A.h[1] = *(const v8h*)(ap + 16);
      const _Float16* bp = kb + (size_t)keycol * EE + e0 + kbB;
      Bf.h[0] = *(const v8h*)bp;
      Bf.h[1] = *(const v8h*)(bp + 16);
      s = wmma16(A.v, Bf.v, s);
    }
    // causal mask + stage to LDS
    for (int r = 0; r < 8; ++r) {
      int row = q0 + r + hi8;
      if (keycol > row) s[r] = -1e30f;
      Sbuf[r + hi8][wave * 16 + lrow] = s[r];
    }
    __syncthreads();

    // online softmax: 16 threads per row, 8 cols each
    {
      int rowl = tid >> 4, coll = tid & 15;
      float sv[8];
      float mloc = -1e30f;
      for (int j = 0; j < 8; ++j) {
        sv[j] = Sbuf[rowl][coll + 16 * j];
        mloc = fmaxf(mloc, sv[j]);
      }
      for (int off = 1; off < 16; off <<= 1)
        mloc = fmaxf(mloc, __shfl_xor(mloc, off, 16));
      float mold = mstate[rowl];
      float mnew = fmaxf(mold, mloc);
      float sc = __expf(mold - mnew);
      float sum = 0.0f;
      for (int j = 0; j < 8; ++j) {
        float p = __expf(sv[j] - mnew);
        sum += p;
        Pbuf[rowl][coll + 16 * j] = (_Float16)p;
      }
      for (int off = 1; off < 16; off <<= 1)
        sum += __shfl_xor(sum, off, 16);
      if (coll == 0) {
        mstate[rowl] = mnew;
        lstate[rowl] = lstate[rowl] * sc + sum;
        scl[rowl] = sc;
      }
    }
    __syncthreads();

    // rescale O accumulators by exp(m_old - m_new)
    {
      v4f s0 = *(const v4f*)&scl[hi8];
      v4f s1 = *(const v4f*)&scl[hi8 + 4];
      for (int nt = 0; nt < 4; ++nt)
        for (int r = 0; r < 4; ++r) {
          oacc[nt][r]     *= s0[r];
          oacc[nt][4 + r] *= s1[r];
        }
    }

    // O += P V  (wave owns E cols [wave*64, wave*64+64))
    for (int ks = 0; ks < 4; ++ks) {
      ABfrag A;
      const _Float16* ap = &Pbuf[lrow][ks * 32 + kbA];
      A.h[0] = *(const v8h*)ap;
      A.h[1] = *(const v8h*)(ap + 16);
      for (int nt = 0; nt < 4; ++nt) {
        int ecol = wave * 64 + nt * 16 + lrow;
        const _Float16* bp = vb + (size_t)ecol * TT + kt0 + ks * 32 + kbB;
        ABfrag Bf;
        Bf.h[0] = *(const v8h*)bp;
        Bf.h[1] = *(const v8h*)(bp + 16);
        oacc[nt] = wmma16(A.v, Bf.v, oacc[nt]);
      }
    }

    // prefetch next key tile (emits global_prefetch_b8)
    if (kt + 1 < nkt) {
      __builtin_prefetch(kb + (size_t)(kt0 + 128 + (tid >> 1)) * EE, 0, 0);
    }
  }
  __syncthreads();

  // normalize by 1/l and store enc f16 [B,T,E]
  v4f l0 = *(const v4f*)&lstate[hi8];
  v4f l1 = *(const v4f*)&lstate[hi8 + 4];
  float inv[8];
  for (int r = 0; r < 4; ++r) {
    inv[r]     = 1.0f / l0[r];
    inv[4 + r] = 1.0f / l1[r];
  }
  for (int nt = 0; nt < 4; ++nt) {
    int ecol = wave * 64 + nt * 16 + lrow;
    for (int r = 0; r < 8; ++r) {
      int row = q0 + r + hi8;
      ench[((size_t)b * TT + row) * EE + ecol] = (_Float16)(oacc[nt][r] * inv[r]);
    }
  }
}

// ---------------------------------------------------------------------------
// Kernel 3: y = enc @ W2^T + b2 + residual; LayerNorm over C; f32 out.
// Block = 16 t-rows x full C=512; 8 waves x 64 cols. grid = (T/16, B).
// ---------------------------------------------------------------------------
__global__ __launch_bounds__(256) void proj_ln(
    const _Float16* __restrict__ ench, const _Float16* __restrict__ W2h,
    const float* __restrict__ b2, const float* __restrict__ x,
    const float* __restrict__ gamma, const float* __restrict__ beta,
    float* __restrict__ out) {
  __shared__ __align__(16) float ybuf[16][516];
  __shared__ float mu_s[16], rs_s[16];

  const int q0 = blockIdx.x * 16;
  const int b  = blockIdx.y;
  const int tid  = threadIdx.x;
  const int lane = tid & 31;
  const int wave = tid >> 5;
  const int lrow = lane & 15;
  const int kbA = (lane < 16) ? 0 : 8;
  const int kbB = (lane < 16) ? 0 : 8;
  const int hi8 = (lane < 16) ? 0 : 8;
  const int cb = wave * 64;

  const _Float16* eb = ench + (size_t)b * TT * EE;
  const float* xb = x + (size_t)b * CC * TT;

  v8f acc[4];
  for (int i = 0; i < 4; ++i) acc[i] = (v8f)0.0f;

  for (int e0 = 0; e0 < EE; e0 += 32) {
    ABfrag A;
    const _Float16* ap = eb + (size_t)(q0 + lrow) * EE + e0 + kbA;
    A.h[0] = *(const v8h*)ap;
    A.h[1] = *(const v8h*)(ap + 16);
    for (int nt = 0; nt < 4; ++nt) {
      int c = cb + nt * 16 + lrow;
      const _Float16* bp = W2h + (size_t)c * EE + e0 + kbB;
      ABfrag Bf;
      Bf.h[0] = *(const v8h*)bp;
      Bf.h[1] = *(const v8h*)(bp + 16);
      acc[nt] = wmma16(A.v, Bf.v, acc[nt]);
    }
  }

  // + bias + residual -> LDS
  for (int nt = 0; nt < 4; ++nt) {
    int c = cb + nt * 16 + lrow;
    float bval = b2[c];
    for (int r = 0; r < 8; ++r) {
      int row = r + hi8;
      int t = q0 + row;
      ybuf[row][c] = acc[nt][r] + bval + xb[(size_t)c * TT + t];
    }
  }
  __syncthreads();

  // LayerNorm stats: 16 threads per row, 32 cols each, shfl-reduce width 16
  {
    int rowl = tid >> 4, coll = tid & 15;
    float sum = 0.0f, sq = 0.0f;
    for (int j = 0; j < 32; ++j) {
      float v = ybuf[rowl][coll + 16 * j];
      sum += v;
      sq += v * v;
    }
    for (int off = 1; off < 16; off <<= 1) {
      sum += __shfl_xor(sum, off, 16);
      sq  += __shfl_xor(sq, off, 16);
    }
    if (coll == 0) {
      float mu = sum * (1.0f / CC);
      float var = sq * (1.0f / CC) - mu * mu;
      mu_s[rowl] = mu;
      rs_s[rowl] = rsqrtf(var + 1e-5f);
    }
  }
  __syncthreads();

  // coalesced normalize + affine + store
  for (int j = 0; j < 32; ++j) {
    int idx = tid + 256 * j;           // 16*512 = 8192 elements
    int row = idx >> 9;
    int c = idx & 511;
    float val = (ybuf[row][c] - mu_s[row]) * rs_s[row] * gamma[c] + beta[c];
    out[((size_t)b * TT + q0 + row) * CC + c] = val;
  }
}

// ---------------------------------------------------------------------------
extern "C" void kernel_launch(void* const* d_in, const int* in_sizes, int n_in,
                              void* d_out, int out_size, void* d_ws,
                              size_t ws_size, hipStream_t stream) {
  const float* x     = (const float*)d_in[0];
  const float* Wq    = (const float*)d_in[1];
  const float* bq    = (const float*)d_in[2];
  const float* Wk    = (const float*)d_in[3];
  const float* bk    = (const float*)d_in[4];
  const float* Wv    = (const float*)d_in[5];
  const float* bv    = (const float*)d_in[6];
  const float* W2    = (const float*)d_in[7];
  const float* b2    = (const float*)d_in[8];
  const float* gamma = (const float*)d_in[9];
  const float* beta  = (const float*)d_in[10];
  float* out = (float*)d_out;

  // ws layout (f16): Wh[4*512*512] | qh | kh | vth(E,T) | ench  (~66 MB)
  _Float16* Wh  = (_Float16*)d_ws;
  _Float16* qh  = Wh + (size_t)4 * EE * CC;
  _Float16* kh  = qh + (size_t)BB * TT * EE;
  _Float16* vth = kh + (size_t)BB * TT * EE;
  _Float16* ench = vth + (size_t)BB * TT * EE;

  cvt_weights<<<4096, 256, 0, stream>>>(Wq, Wk, Wv, W2, Wh);

  dim3 g1(TT / 128, BB, 24);   // 3 weights * (E/64) chunks
  qkv_gemm<<<g1, 256, 0, stream>>>(x, Wh, bq, bk, bv, qh, kh, vth);

  dim3 g2(TT / 16, BB);
  attn_flash<<<g2, 256, 0, stream>>>(qh, kh, vth, ench);

  proj_ln<<<g2, 256, 0, stream>>>(ench, Wh + (size_t)3 * EE * CC, b2, x, gamma,
                                  beta, out);
}
